// SparseSegNet_51299089383597
// MI455X (gfx1250) — matrix operand order repaired and checked by reading, not compiled, not run on hardware
//
#include <hip/hip_runtime.h>

typedef _Float16 h16;
typedef __attribute__((ext_vector_type(8)))  _Float16 v8h;
typedef __attribute__((ext_vector_type(16))) _Float16 v16h;
typedef __attribute__((ext_vector_type(8)))  float    v8f;
typedef __attribute__((ext_vector_type(4)))  int      v4i;

#define CDIM 128
#define NCLS 19
#define LDSW 136   // padded f16 row stride (64-bank conflict avoidance)

// ---- CDNA5 async global->LDS copy path (ASYNCcnt), guarded so the build
// ---- falls back to VGPR-staged copies if the toolchain lacks the builtins.
#if defined(__has_builtin)
#  if __has_builtin(__builtin_amdgcn_global_load_async_to_lds_b128) && \
      __has_builtin(__builtin_amdgcn_s_wait_asynccnt)
#    define USE_ASYNC_LDS 1
#  endif
#endif
#ifndef USE_ASYNC_LDS
#  define USE_ASYNC_LDS 0
#endif

#if USE_ASYNC_LDS
typedef __attribute__((address_space(1))) v4i gv4i;   // global int4 (b128 payload)
typedef __attribute__((address_space(3))) v4i lv4i;   // LDS int4
#endif

static __device__ __forceinline__ v16h cat16(v8h a, v8h b) {
  return __builtin_shufflevector(a, b, 0,1,2,3,4,5,6,7,8,9,10,11,12,13,14,15);
}

// ---------------------------------------------------------------------------
// WMMA sparse-conv / dense-GEMM kernel.
// out[i,co] = bias[co] + sum_k sum_ci inP[nbr[k,i], ci] * Wt[k, co, ci]
// inP is (Ntot+1) x 128 f16 with row Ntot == zeros (missing neighbor).
// nbr == nullptr -> identity gather (dense GEMM), numK == 1.
// Block: 256 threads (8 waves). Tile: 64 rows x 128 cols.
// ---------------------------------------------------------------------------
__global__ __launch_bounds__(256) void k_conv_wmma(
    const h16* __restrict__ inP, const h16* __restrict__ Wt,
    const float* __restrict__ bias, const int* __restrict__ nbr,
    float* __restrict__ out, int Ntot, int numK)
{
  __shared__ h16 sA[64 * LDSW];
  __shared__ h16 sB[CDIM * LDSW];
  const int t  = threadIdx.x;
  const int i0 = blockIdx.x * 64;
  const int w = t >> 5, lane = t & 31;
  const int mt = w & 3;            // M-tile 0..3 (16 rows each)
  const int nb = (w >> 2) * 64;    // N base 0 or 64 (4 tiles of 16)
  const int lm = lane & 15, hl = lane >> 4;
  v8f acc[4] = {v8f{}, v8f{}, v8f{}, v8f{}};

  for (int k = 0; k < numK; ++k) {
    __syncthreads();
    // gather A tile: 64 rows x 128 f16  (16B chunks per thread)
#pragma unroll
    for (int it = 0; it < 4; ++it) {
      int lin = it * 256 + t;
      int row = lin >> 4, ch = (lin & 15) << 3;
      int gi = i0 + row;
      int idx = (gi < Ntot) ? (nbr ? nbr[(size_t)k * Ntot + gi] : gi) : Ntot;
#if USE_ASYNC_LDS
      __builtin_amdgcn_global_load_async_to_lds_b128(
          (gv4i*)(inP + (size_t)idx * CDIM + ch),
          (lv4i*)(sA + row * LDSW + ch), 0, 0);
#else
      *(v8h*)(sA + row * LDSW + ch) = *(const v8h*)(inP + (size_t)idx * CDIM + ch);
#endif
    }
    // B tile: W[k] transposed (co-major), 128 x 128 f16
#pragma unroll
    for (int it = 0; it < 8; ++it) {
      int lin = it * 256 + t;
      int row = lin >> 4, ch = (lin & 15) << 3;
#if USE_ASYNC_LDS
      __builtin_amdgcn_global_load_async_to_lds_b128(
          (gv4i*)(Wt + ((size_t)k * CDIM + row) * CDIM + ch),
          (lv4i*)(sB + row * LDSW + ch), 0, 0);
#else
      *(v8h*)(sB + row * LDSW + ch) =
          *(const v8h*)(Wt + ((size_t)k * CDIM + row) * CDIM + ch);
#endif
    }
#if USE_ASYNC_LDS
    __builtin_amdgcn_s_wait_asynccnt(0);   // drain this wave's async copies
#endif
    __syncthreads();
#pragma unroll
    for (int kc = 0; kc < 4; ++kc) {
      const int kb = kc * 32;
      // A-frag (16x32 f16): lanes 0-15 hold K kb..kb+7 & kb+16..kb+23
      const h16* pa = sA + (mt * 16 + lm) * LDSW + kb + (hl << 3);
      v16h a = cat16(*(const v8h*)pa, *(const v8h*)(pa + 16));
#pragma unroll
      for (int j = 0; j < 4; ++j) {
        // B-frag (32x16): lane holds column co = nb+j*16+lm, K kb+8h..+7 & kb+16+8h..+7
        const h16* pb = sB + (nb + j * 16 + lm) * LDSW + kb + (hl << 3);
        v16h b = cat16(*(const v8h*)pb, *(const v8h*)(pb + 16));
        acc[j] = __builtin_amdgcn_wmma_f32_16x16x32_f16(
            false, a, false, b, (short)0, acc[j], false, false);
      }
    }
  }
  // epilogue: C layout lanes 0-15 -> M=e, lanes 16-31 -> M=e+8
#pragma unroll
  for (int j = 0; j < 4; ++j) {
    int col = nb + j * 16 + lm;
    float bc = bias[col];
#pragma unroll
    for (int e = 0; e < 8; ++e) {
      int row = i0 + mt * 16 + (hl << 3) + e;
      if (row < Ntot) out[(size_t)row * CDIM + col] = acc[j][e] + bc;
    }
  }
}

// ---------------------------------------------------------------------------
// scalar stem conv: 4 -> 128, reduction 27*4. One block (128 thr) per point.
// ---------------------------------------------------------------------------
__global__ __launch_bounds__(128) void k_stem(
    const float* __restrict__ feat, const int* __restrict__ nbr,
    const float* __restrict__ Ws, const float* __restrict__ bs,
    float* __restrict__ out, int Ntot)
{
  int i = blockIdx.x;
  int co = threadIdx.x;
  float acc = bs[co];
  for (int k = 0; k < 27; ++k) {
    int idx = nbr[(size_t)k * Ntot + i];
    if (idx < Ntot) {
      const float* f = feat + (size_t)idx * 4;
      const float* wk = Ws + ((size_t)k * 4) * CDIM + co;
      acc += f[0] * wk[0] + f[1] * wk[CDIM] + f[2] * wk[2 * CDIM] + f[3] * wk[3 * CDIM];
    }
  }
  out[(size_t)i * CDIM + co] = acc;
}

// per-channel sum / sumsq over N rows -> stats[0..127]=sum, stats[128..255]=sumsq
__global__ __launch_bounds__(256) void k_stats(
    const float* __restrict__ x, float* __restrict__ stats, int Ntot)
{
  __shared__ float ss[256], sq[256];
  int t = threadIdx.x;
  int c = t & 127, h = t >> 7;
  long long base = (long long)blockIdx.x * 512;
  int rows = (int)((Ntot - base < 512) ? (Ntot - base) : 512);
  float ls = 0.f, lq = 0.f;
  for (int r = h; r < rows; r += 2) {
    float v = x[(base + r) * CDIM + c];
    ls += v; lq += v * v;
  }
  ss[t] = ls; sq[t] = lq;
  __syncthreads();
  if (t < 128) {
    atomicAdd(&stats[c],       ss[t] + ss[t + 128]);
    atomicAdd(&stats[128 + c], sq[t] + sq[t + 128]);
  }
}

__global__ void k_zero(float* p, int n) {
  int i = blockIdx.x * blockDim.x + threadIdx.x;
  if (i < n) p[i] = 0.f;
}

static __device__ __forceinline__ void meanrstd(const float* stats, int c, int Ntot,
                                                float& m, float& r) {
  m = stats[c] / (float)Ntot;
  float var = stats[128 + c] / (float)Ntot - m * m;
  r = rsqrtf(var + 1e-5f);
}

// y = relu(norm(x)) as f16 into padded buffer (row Ntot zeroed)
__global__ void k_norm_relu_h16(const float* __restrict__ x, const float* __restrict__ stats,
                                h16* __restrict__ outH, int Ntot)
{
  size_t idx = (size_t)blockIdx.x * 256 + threadIdx.x;
  size_t total = (size_t)(Ntot + 1) * CDIM;
  if (idx >= total) return;
  int row = (int)(idx >> 7), c = (int)(idx & 127);
  if (row >= Ntot) { outH[idx] = (h16)0.f; return; }
  float m, r; meanrstd(stats, c, Ntot, m, r);
  float v = (x[idx] - m) * r;
  outH[idx] = (h16)fmaxf(v, 0.f);
}

// x2 = relu(x1 + norm(c2))   (x1 recovered from f16 buffer)
__global__ void k_resid_relu(const float* __restrict__ c2, const h16* __restrict__ x1h,
                             const float* __restrict__ stats, float* __restrict__ outF, int Ntot)
{
  size_t idx = (size_t)blockIdx.x * 256 + threadIdx.x;
  if (idx >= (size_t)Ntot * CDIM) return;
  int c = (int)(idx & 127);
  float m, r; meanrstd(stats, c, Ntot, m, r);
  outF[idx] = fmaxf((float)x1h[idx] + (c2[idx] - m) * r, 0.f);
}

// x3 = relu(norm(x2)) -> f16 padded AND f32 copy
__global__ void k_norm_relu_both(const float* __restrict__ x2, const float* __restrict__ stats,
                                 h16* __restrict__ x3h, float* __restrict__ x3f, int Ntot)
{
  size_t idx = (size_t)blockIdx.x * 256 + threadIdx.x;
  size_t total = (size_t)(Ntot + 1) * CDIM;
  if (idx >= total) return;
  int row = (int)(idx >> 7), c = (int)(idx & 127);
  if (row >= Ntot) { x3h[idx] = (h16)0.f; return; }
  float m, r; meanrstd(stats, c, Ntot, m, r);
  float v = fmaxf((x2[idx] - m) * r, 0.f);
  x3h[idx] = (h16)v;
  x3f[idx] = v;
}

// se = leakyrelu(norm(sepre)); write se to d_out section; x4 = se + x3
__global__ void k_se_final(const float* __restrict__ sepre, const float* __restrict__ stats,
                           const float* __restrict__ x3f, float* __restrict__ seOut,
                           float* __restrict__ x4, int Ntot)
{
  size_t idx = (size_t)blockIdx.x * 256 + threadIdx.x;
  if (idx >= (size_t)Ntot * CDIM) return;
  int c = (int)(idx & 127);
  float m, r; meanrstd(stats, c, Ntot, m, r);
  float v = (sepre[idx] - m) * r;
  v = (v >= 0.f) ? v : 0.2f * v;
  seOut[idx] = v;
  x4[idx] = v + x3f[idx];
}

// logits = x4 @ Wlin + blin ; weighted CE partial sums via atomics
__global__ __launch_bounds__(256) void k_logits_loss(
    const float* __restrict__ x, const float* __restrict__ Wl, const float* __restrict__ bl,
    const int* __restrict__ label, const float* __restrict__ wc,
    float* __restrict__ logits, float* __restrict__ acc, int Ntot)
{
  __shared__ float sw[2];
  if (threadIdx.x == 0) { sw[0] = 0.f; sw[1] = 0.f; }
  __syncthreads();
  int i = blockIdx.x * 256 + threadIdx.x;
  if (i < Ntot) {
    float lg[NCLS];
#pragma unroll
    for (int j = 0; j < NCLS; ++j) lg[j] = bl[j];
    for (int c = 0; c < CDIM; ++c) {
      float xv = x[(size_t)i * CDIM + c];
#pragma unroll
      for (int j = 0; j < NCLS; ++j) lg[j] += xv * Wl[c * NCLS + j];
    }
    float mx = lg[0];
#pragma unroll
    for (int j = 1; j < NCLS; ++j) mx = fmaxf(mx, lg[j]);
    float se = 0.f;
#pragma unroll
    for (int j = 0; j < NCLS; ++j) se += __expf(lg[j] - mx);
    float lse = mx + __logf(se);
#pragma unroll
    for (int j = 0; j < NCLS; ++j) logits[(size_t)i * NCLS + j] = lg[j];
    int lb = label[i];
    float nll = lse - lg[lb];
    float ww = wc[lb];
    atomicAdd(&sw[0], ww * nll);
    atomicAdd(&sw[1], ww);
  }
  __syncthreads();
  if (threadIdx.x == 0) { atomicAdd(&acc[0], sw[0]); atomicAdd(&acc[1], sw[1]); }
}

__global__ void k_finalize(const float* acc, float* loss) { loss[0] = acc[0] / acc[1]; }

// W [K][ci][co] f32 -> Wt [K][co][ci] f16
__global__ void k_prep_w(const float* __restrict__ W, h16* __restrict__ Wt, int K) {
  int idx = blockIdx.x * 256 + threadIdx.x;
  if (idx >= K * CDIM * CDIM) return;
  int k = idx >> 14, rem = idx & 16383;
  int ci = rem >> 7, co = rem & 127;
  Wt[((size_t)k * CDIM + co) * CDIM + ci] = (h16)W[idx];
}

extern "C" void kernel_launch(void* const* d_in, const int* in_sizes, int n_in,
                              void* d_out, int out_size, void* d_ws, size_t ws_size,
                              hipStream_t stream) {
  const float* feat  = (const float*)d_in[1];
  const int*   label = (const int*)d_in[2];
  const float* wcls  = (const float*)d_in[3];
  const int*   nbr   = (const int*)d_in[4];
  const float* Wstem = (const float*)d_in[5];
  const float* bstem = (const float*)d_in[6];
  const float* W1    = (const float*)d_in[7];
  const float* b1    = (const float*)d_in[8];
  const float* W2    = (const float*)d_in[9];
  const float* b2    = (const float*)d_in[10];
  const float* Wse   = (const float*)d_in[11];
  const float* bse   = (const float*)d_in[12];
  const float* Wlin  = (const float*)d_in[13];
  const float* blin  = (const float*)d_in[14];
  const int Ntot = in_sizes[2];   // label count == N

  // workspace carve-up (256B aligned)
  char* ws = (char*)d_ws;
  size_t off = 0;
  auto carve = [&](size_t bytes) { void* p = ws + off; off += (bytes + 255) & ~(size_t)255; return p; };
  float* A    = (float*)carve((size_t)Ntot * CDIM * 4);
  float* B    = (float*)carve((size_t)Ntot * CDIM * 4);
  h16*   H1   = (h16*)carve((size_t)(Ntot + 1) * CDIM * 2);
  h16*   H2   = (h16*)carve((size_t)(Ntot + 1) * CDIM * 2);
  h16*   W1t  = (h16*)carve((size_t)27 * CDIM * CDIM * 2);
  h16*   W2t  = (h16*)carve((size_t)27 * CDIM * CDIM * 2);
  h16*   WseT = (h16*)carve((size_t)CDIM * CDIM * 2);
  float* stats = (float*)carve(272 * 4);     // [0..127]=sum [128..255]=sumsq [256..257]=loss acc
  float* acc   = stats + 256;

  float* outLogits = (float*)d_out;
  float* outSe     = outLogits + (size_t)Ntot * NCLS;
  float* outLoss   = outSe + (size_t)Ntot * CDIM;

  const int gConv  = (Ntot + 63) / 64;
  const int gStats = (Ntot + 511) / 512;
  const int gPad   = (int)(((size_t)(Ntot + 1) * CDIM + 255) / 256);
  const int gN     = (int)(((size_t)Ntot * CDIM + 255) / 256);
  const int gPts   = (Ntot + 255) / 256;

  // weight prep
  k_prep_w<<<(27 * CDIM * CDIM + 255) / 256, 256, 0, stream>>>(W1, W1t, 27);
  k_prep_w<<<(27 * CDIM * CDIM + 255) / 256, 256, 0, stream>>>(W2, W2t, 27);
  k_prep_w<<<(CDIM * CDIM + 255) / 256, 256, 0, stream>>>(Wse, WseT, 1);

  // stem + inorm + relu
  k_stem<<<Ntot, 128, 0, stream>>>(feat, nbr, Wstem, bstem, A, Ntot);
  k_zero<<<2, 256, 0, stream>>>(stats, 258);
  k_stats<<<gStats, 256, 0, stream>>>(A, stats, Ntot);
  k_norm_relu_h16<<<gPad, 256, 0, stream>>>(A, stats, H1, Ntot);

  // conv1 + inorm + relu
  k_conv_wmma<<<gConv, 256, 0, stream>>>(H1, W1t, b1, nbr, A, Ntot, 27);
  k_zero<<<2, 256, 0, stream>>>(stats, 258);
  k_stats<<<gStats, 256, 0, stream>>>(A, stats, Ntot);
  k_norm_relu_h16<<<gPad, 256, 0, stream>>>(A, stats, H2, Ntot);

  // conv2 + inorm ; residual + relu
  k_conv_wmma<<<gConv, 256, 0, stream>>>(H2, W2t, b2, nbr, A, Ntot, 27);
  k_zero<<<2, 256, 0, stream>>>(stats, 258);
  k_stats<<<gStats, 256, 0, stream>>>(A, stats, Ntot);
  k_resid_relu<<<gN, 256, 0, stream>>>(A, H1, stats, B, Ntot);

  // trailing inorm + relu -> x3 (f16 + f32)
  k_zero<<<2, 256, 0, stream>>>(stats, 258);
  k_stats<<<gStats, 256, 0, stream>>>(B, stats, Ntot);
  k_norm_relu_both<<<gPad, 256, 0, stream>>>(B, stats, H2, A, Ntot);

  // shape embedding: dense GEMM (numK=1, identity gather) + inorm + leaky + add
  k_conv_wmma<<<gConv, 256, 0, stream>>>(H2, WseT, bse, nullptr, B, Ntot, 1);
  k_zero<<<2, 256, 0, stream>>>(stats, 258);
  k_stats<<<gStats, 256, 0, stream>>>(B, stats, Ntot);
  k_se_final<<<gN, 256, 0, stream>>>(B, stats, A, outSe, B, Ntot);

  // logits + weighted CE
  k_logits_loss<<<gPts, 256, 0, stream>>>(B, Wlin, blin, label, wcls, outLogits, acc, Ntot);
  k_finalize<<<1, 1, 0, stream>>>(acc, outLoss);
}
